// SparseHilbertSSM_61881888801260
// MI455X (gfx1250) — compile-verified, matrix operand order checked by reading, not run on hardware
//
#include <hip/hip_runtime.h>
#include <cstdint>
#include <cstddef>

// ---------------- model constants ----------------
#define LSEQ    16384
#define ENC     64
#define DMODEL  256
#define DSTATE  64
#define DCONV   4
#define NLAYERS 3
#define DIN     512
#define HEADDIM 64
#define NHEADS  8
#define DXBC    640
#define DPROJ   1160
#define NDIR    6
#define QCH     64
#define NCHUNK  (LSEQ / QCH)       // 256
#define EPSV    1e-5f

// ---------------- WMMA plumbing (gfx1250, wave32) ----------------
typedef __attribute__((ext_vector_type(16))) __bf16 v16bf;
typedef __attribute__((ext_vector_type(8)))  __bf16 v8bf;
typedef __attribute__((ext_vector_type(8)))  float  v8f;

__device__ __forceinline__ v8f v8f_zero() {
  v8f z = {0.f, 0.f, 0.f, 0.f, 0.f, 0.f, 0.f, 0.f};
  return z;
}

// Per-lane fragment: two contiguous 8x bf16 runs at p[0..7] and p[16..23].
// Matches the documented 16-bit A/B operand layout for 16x16x32:
//   lanes 0-15:  k = 0..7  and 16..23 ;  lanes 16-31: k = 8..15 and 24..31
// (caller pre-applies kb = (lane>>4)*8 into p)
__device__ __forceinline__ v16bf frag_ld(const __bf16* p) {
  v16bf r;
#pragma unroll
  for (int i = 0; i < 8; i++) { r[i] = p[i]; r[i + 8] = p[i + 16]; }
  return r;
}

__device__ __forceinline__ v8f wmma_bf(v16bf a, v16bf b, v8f c) {
  // 8 args: (neg_a, A, neg_b, B, c_mod, C, reuse_a, reuse_b)
  return __builtin_amdgcn_wmma_f32_16x16x32_bf16(false, a, false, b, (short)0, c,
                                                 false, false);
}

// ---------------- small math ----------------
__device__ __forceinline__ float geluf(float x) {
  return 0.5f * x * (1.0f + erff(x * 0.7071067811865475f));
}
__device__ __forceinline__ float siluf(float x) { return x / (1.0f + expf(-x)); }
__device__ __forceinline__ float softplusf(float x) {
  return (x > 20.f) ? x : log1pf(expf(x));
}
__device__ __forceinline__ float block_sum256(float v, float* red) {
  int tid = threadIdx.x;
  red[tid] = v;
  __syncthreads();
  for (int s = 128; s > 0; s >>= 1) {
    if (tid < s) red[tid] += red[tid + s];
    __syncthreads();
  }
  float r = red[0];
  __syncthreads();
  return r;
}

__constant__ int c_ord[NDIR][3] = {{0,1,2},{0,2,1},{1,0,2},{1,2,0},{2,0,1},{2,1,0}};

// ---------------- utility kernels ----------------
__global__ void __launch_bounds__(256) cvt_bf16_kernel(const float* __restrict__ s,
                                                       __bf16* __restrict__ d, int n) {
  int i = blockIdx.x * 256 + threadIdx.x;
  if (i < n) d[i] = (__bf16)s[i];
}

__global__ void __launch_bounds__(256) concat_embed_kernel(const float* __restrict__ re,
                                                           const float* __restrict__ im,
                                                           __bf16* __restrict__ out) {
  int i = blockIdx.x * 256 + threadIdx.x;   // < LSEQ*128
  int l = i >> 7, c = i & 127;
  float v = (c < ENC) ? re[l * ENC + c] : im[l * ENC + (c - ENC)];
  out[i] = (__bf16)v;
}

// ---------------- generic bf16 GEMM: C[M,N] (+)= gelu(A[M,K]@B[K,N] + bias) ----
#define GF_BIAS 1
#define GF_GELU 2
#define GF_ACC  4

__global__ void __launch_bounds__(256) gemm_bf16_kernel(
    const __bf16* __restrict__ A, const __bf16* __restrict__ B,
    const float* __restrict__ bias, float* __restrict__ C,
    int M, int N, int K, int flags) {
  __shared__ __bf16 sB[64][264];       // B slab transposed: [n_local][k_local]
  const int tid  = threadIdx.x;
  const int lane = tid & 31, wav = tid >> 5;
  const int sub  = lane & 15, hb = (lane >> 4) * 8;
  const int mtile = blockIdx.x * 128;
  const int ntile = blockIdx.y * 64;

  v8f acc[4] = {v8f_zero(), v8f_zero(), v8f_zero(), v8f_zero()};

  for (int ks = 0; ks < K; ks += 256) {
    int kcur = K - ks; if (kcur > 256) kcur = 256;
    if (ks + 256 < K)  // prefetch next B slab -> global_prefetch_b8
      __builtin_prefetch(B + (size_t)(ks + 256) * N + ntile + (tid & 63), 0, 1);
    for (int idx = tid; idx < kcur * 64; idx += 256) {
      int kk = idx >> 6, nn = idx & 63;
      int gn = ntile + nn;
      __bf16 v = (__bf16)0.0f;
      if (gn < N) v = B[(size_t)(ks + kk) * N + gn];
      sB[nn][kk] = v;
    }
    __syncthreads();
    int steps = kcur >> 5;
    for (int kt = 0; kt < steps; kt++) {
      const __bf16* ap = A + (size_t)(mtile + wav * 16 + sub) * K + ks + kt * 32 + hb;
      v16bf af = frag_ld(ap);
#pragma unroll
      for (int nf = 0; nf < 4; nf++) {
        v16bf bf = frag_ld(&sB[nf * 16 + sub][kt * 32 + hb]);
        acc[nf] = wmma_bf(af, bf, acc[nf]);
      }
    }
    __syncthreads();
  }
#pragma unroll
  for (int nf = 0; nf < 4; nf++) {
    int n = ntile + nf * 16 + sub;
    if (n >= N) continue;
    float bv = (flags & GF_BIAS) ? bias[n] : 0.0f;
#pragma unroll
    for (int i = 0; i < 8; i++) {
      int m = mtile + wav * 16 + hb + i;
      size_t o = (size_t)m * N + n;
      float v = acc[nf][i] + bv;
      if (flags & GF_ACC) v += C[o];
      if (flags & GF_GELU) v = geluf(v);
      C[o] = v;
    }
  }
}

// ---------------- row LayerNorm over 256 cols ----------------
__global__ void __launch_bounds__(256) ln256_kernel(
    const float* __restrict__ src, const float* __restrict__ w,
    const float* __restrict__ b, float* __restrict__ outf,
    __bf16* __restrict__ outb, int dogelu) {
  __shared__ float red[256];
  int r = blockIdx.x, tid = threadIdx.x;
  float x = src[(size_t)r * DMODEL + tid];
  float mu = block_sum256(x, red) * (1.0f / DMODEL);
  float dv = x - mu;
  float var = block_sum256(dv * dv, red) * (1.0f / DMODEL);
  float y = dv * rsqrtf(var + EPSV) * w[tid] + b[tid];
  if (dogelu) y = geluf(y);
  if (outf) outf[(size_t)r * DMODEL + tid] = y;
  if (outb) outb[(size_t)r * DMODEL + tid] = (__bf16)y;
}

// ---------------- lexsort: keys, ranks, gather ----------------
__global__ void __launch_bounds__(256) key_kernel(const int* __restrict__ coords,
                                                  unsigned* __restrict__ keys) {
  int i = blockIdx.x * 256 + threadIdx.x;   // < NDIR*LSEQ
  int d = i / LSEQ, t = i % LSEQ;
  const int* c = coords + (size_t)t * 4;
  unsigned k = ((unsigned)c[c_ord[d][0]] << 14) |
               ((unsigned)c[c_ord[d][1]] << 7) |
               (unsigned)c[c_ord[d][2]];
  keys[i] = k;
}

__global__ void __launch_bounds__(256) rank_kernel(const unsigned* __restrict__ keys,
                                                   int* __restrict__ inv,
                                                   int* __restrict__ perm) {
  __shared__ unsigned sk[256];
  int d = blockIdx.y;
  int i = blockIdx.x * 256 + threadIdx.x;
  unsigned ki = keys[(size_t)d * LSEQ + i];
  int r = 0;
  for (int base = 0; base < LSEQ; base += 256) {
    sk[threadIdx.x] = keys[(size_t)d * LSEQ + base + threadIdx.x];
    __syncthreads();
#pragma unroll 8
    for (int j = 0; j < 256; j++) {
      unsigned kj = sk[j];
      int jj = base + j;
      r += (kj < ki) || (kj == ki && jj < i);
    }
    __syncthreads();
  }
  inv[(size_t)d * LSEQ + i]  = r;
  perm[(size_t)d * LSEQ + r] = i;
}

__global__ void __launch_bounds__(256) gather_kernel(const float* __restrict__ h,
                                                     const int* __restrict__ perm,
                                                     float* __restrict__ xs) {
  size_t i = (size_t)blockIdx.x * 256 + threadIdx.x;   // < NDIR*LSEQ*DMODEL
  int c = (int)(i & (DMODEL - 1));
  size_t row = i >> 8;                 // d*LSEQ + j
  int src = perm[row];
  xs[i] = h[(size_t)src * DMODEL + c];
}

// ---------------- Mamba2 elementwise pieces ----------------
__global__ void __launch_bounds__(256) dt_kernel(const float* __restrict__ zx,
                                                 const float* __restrict__ dtb,
                                                 float* __restrict__ dtp) {
  int i = blockIdx.x * 256 + threadIdx.x;   // < LSEQ*NHEADS
  int t = i >> 3, hh = i & 7;
  float v = zx[(size_t)t * DPROJ + (DIN + DXBC) + hh] + dtb[hh];
  dtp[i] = softplusf(v);
}

__global__ void __launch_bounds__(256) conv_kernel(const float* __restrict__ zx,
                                                   const float* __restrict__ cw,
                                                   const float* __restrict__ cb,
                                                   float* __restrict__ xconv) {
  size_t i = (size_t)blockIdx.x * 256 + threadIdx.x;   // < LSEQ*DXBC
  int t = (int)(i / DXBC), c = (int)(i % DXBC);
  float a = cb[c];
#pragma unroll
  for (int j = 0; j < DCONV; j++) {
    int tt = t - (DCONV - 1) + j;
    if (tt >= 0) a += cw[c * DCONV + j] * zx[(size_t)tt * DPROJ + DIN + c];
  }
  xconv[i] = siluf(a);
}

// ---------------- chunked SSD: intra-chunk (WMMA heavy) ----------------
__global__ void __launch_bounds__(256) ssm_intra_kernel(
    const float* __restrict__ xconv, const float* __restrict__ dtp,
    const float* __restrict__ A_log, const float* __restrict__ Dskip,
    float* __restrict__ y, float* __restrict__ locals,
    float* __restrict__ cdecay, float* __restrict__ eA) {
  __shared__ __bf16 sC[64][72];   // C chunk [t][s]; later reused as G[s][t]
  __shared__ __bf16 sB[64][72];   // B chunk [t][s]; later reused as M[t][s]
  __shared__ __bf16 sXt[64][72];  // X transposed [p][t]
  __shared__ float  sCB[64][66];  // C@B^T f32
  __shared__ float  sDt[64];
  __shared__ float  sEl[64];      // A * cumsum(dt)

  const int ci = blockIdx.x, hh = blockIdx.y;
  const int tid = threadIdx.x;
  const int lane = tid & 31, wav = tid >> 5;
  const int sub = lane & 15, hb = (lane >> 4) * 8;

  for (int idx = tid; idx < 64 * 64; idx += 256) {
    int t = idx >> 6, s = idx & 63;
    const float* row = xconv + (size_t)(ci * 64 + t) * DXBC;
    sC[t][s]  = (__bf16)row[DIN + DSTATE + s];
    sB[t][s]  = (__bf16)row[DIN + s];
    sXt[s][t] = (__bf16)row[hh * 64 + s];
  }
  if (tid < 64) sDt[tid] = dtp[(size_t)(ci * 64 + tid) * NHEADS + hh];
  __syncthreads();
  if (tid == 0) {
    float Ah = -expf(A_log[hh]);
    float c = 0.f;
    for (int t = 0; t < 64; t++) {
      c += sDt[t];
      sEl[t] = c * Ah;
      eA[(size_t)(ci * 64 + t) * NHEADS + hh] = sEl[t];
    }
    cdecay[ci * NHEADS + hh] = expf(sEl[63]);
  }
  __syncthreads();

  // CBt[t][s2] = sum_s C[t][s] * B[s2][s]
  for (int r = 0; r < 2; r++) {
    int tt = wav * 2 + r, tm = tt >> 2, tn = tt & 3;
    v8f acc = v8f_zero();
#pragma unroll
    for (int k2 = 0; k2 < 2; k2++) {
      v16bf a = frag_ld(&sC[tm * 16 + sub][k2 * 32 + hb]);
      v16bf b = frag_ld(&sB[tn * 16 + sub][k2 * 32 + hb]);
      acc = wmma_bf(a, b, acc);
    }
#pragma unroll
    for (int i = 0; i < 8; i++) sCB[tm * 16 + hb + i][tn * 16 + sub] = acc[i];
  }
  __syncthreads();
  // G[s][t] = exp(el[63]-el[t]) * dt_t * B[t][s]   (into sC)
  for (int idx = tid; idx < 64 * 64; idx += 256) {
    int t = idx >> 6, s = idx & 63;
    float g = expf(sEl[63] - sEl[t]) * sDt[t] * (float)sB[t][s];
    sC[s][t] = (__bf16)g;
  }
  __syncthreads();
  // M[t][s] = (s<=t) ? exp(el[t]-el[s]) * dt_s * CBt[t][s] : 0   (into sB)
  for (int idx = tid; idx < 64 * 64; idx += 256) {
    int t = idx >> 6, s = idx & 63;
    float m = (s <= t) ? expf(sEl[t] - sEl[s]) * sDt[s] * sCB[t][s] : 0.f;
    sB[t][s] = (__bf16)m;
  }
  __syncthreads();

  float Dsk = Dskip[hh];
  // Y_intra[t][p] = (M @ X)[t][p] + Dskip * x[t][p]
  for (int r = 0; r < 2; r++) {
    int tt = wav * 2 + r, tm = tt >> 2, tn = tt & 3;
    v8f acc = v8f_zero();
#pragma unroll
    for (int k2 = 0; k2 < 2; k2++) {
      v16bf a = frag_ld(&sB[tm * 16 + sub][k2 * 32 + hb]);
      v16bf b = frag_ld(&sXt[tn * 16 + sub][k2 * 32 + hb]);
      acc = wmma_bf(a, b, acc);
    }
    int p = tn * 16 + sub;
#pragma unroll
    for (int i = 0; i < 8; i++) {
      int t = tm * 16 + hb + i;
      y[(size_t)(ci * 64 + t) * DIN + hh * 64 + p] = acc[i] + Dsk * (float)sXt[p][t];
    }
  }
  // local[p][s] = sum_t X^T[p][t] * G[s][t]
  for (int r = 0; r < 2; r++) {
    int tt = wav * 2 + r, tm = tt >> 2, tn = tt & 3;
    v8f acc = v8f_zero();
#pragma unroll
    for (int k2 = 0; k2 < 2; k2++) {
      v16bf a = frag_ld(&sXt[tm * 16 + sub][k2 * 32 + hb]);
      v16bf b = frag_ld(&sC[tn * 16 + sub][k2 * 32 + hb]);
      acc = wmma_bf(a, b, acc);
    }
    float* lb = locals + ((size_t)ci * NHEADS + hh) * 4096;
#pragma unroll
    for (int i = 0; i < 8; i++)
      lb[(tm * 16 + hb + i) * 64 + tn * 16 + sub] = acc[i];
  }
}

// ---------------- chunked SSD: serial state recurrence ----------------
__global__ void __launch_bounds__(256) ssm_state_kernel(
    const float* __restrict__ locals, const float* __restrict__ cdecay,
    float* __restrict__ states) {
  const int hh = blockIdx.x, tid = threadIdx.x;
  float S[16];
#pragma unroll
  for (int j = 0; j < 16; j++) S[j] = 0.f;
  for (int c = 0; c < NCHUNK; c++) {
    size_t base = ((size_t)c * NHEADS + hh) * 4096 + (size_t)tid * 16;
    float dc = cdecay[c * NHEADS + hh];
#pragma unroll
    for (int j = 0; j < 16; j++) {
      states[base + j] = S[j];
      S[j] = S[j] * dc + locals[base + j];
    }
  }
}

// ---------------- chunked SSD: inter-chunk (WMMA) ----------------
__global__ void __launch_bounds__(256) ssm_inter_kernel(
    const float* __restrict__ xconv, const float* __restrict__ states,
    const float* __restrict__ eA, float* __restrict__ y) {
  __shared__ __bf16 sC[64][72];   // C chunk [t][s]
  __shared__ __bf16 sSp[64][72];  // S_prev [p][s]
  __shared__ float  sEl[64];
  const int ci = blockIdx.x, hh = blockIdx.y;
  const int tid = threadIdx.x;
  const int lane = tid & 31, wav = tid >> 5;
  const int sub = lane & 15, hb = (lane >> 4) * 8;

  for (int idx = tid; idx < 64 * 64; idx += 256) {
    int a = idx >> 6, b = idx & 63;
    sC[a][b]  = (__bf16)xconv[(size_t)(ci * 64 + a) * DXBC + DIN + DSTATE + b];
    sSp[a][b] = (__bf16)states[((size_t)ci * NHEADS + hh) * 4096 + idx];
  }
  if (tid < 64) sEl[tid] = eA[(size_t)(ci * 64 + tid) * NHEADS + hh];
  __syncthreads();

  // Y_inter[t][p] = exp(el[t]) * sum_s C[t][s] * Sprev[p][s]
  for (int r = 0; r < 2; r++) {
    int tt = wav * 2 + r, tm = tt >> 2, tn = tt & 3;
    v8f acc = v8f_zero();
#pragma unroll
    for (int k2 = 0; k2 < 2; k2++) {
      v16bf a = frag_ld(&sC[tm * 16 + sub][k2 * 32 + hb]);
      v16bf b = frag_ld(&sSp[tn * 16 + sub][k2 * 32 + hb]);
      acc = wmma_bf(a, b, acc);
    }
    int p = tn * 16 + sub;
#pragma unroll
    for (int i = 0; i < 8; i++) {
      int t = tm * 16 + hb + i;
      y[(size_t)(ci * 64 + t) * DIN + hh * 64 + p] += expf(sEl[t]) * acc[i];
    }
  }
}

// ---------------- gated RMSNorm -> bf16 ----------------
__global__ void __launch_bounds__(256) gate_rms_kernel(
    const float* __restrict__ y, const float* __restrict__ zx,
    const float* __restrict__ rmsw, __bf16* __restrict__ out) {
  __shared__ float red[256];
  int r = blockIdx.x, tid = threadIdx.x;
  float v0 = y[(size_t)r * DIN + tid]       * siluf(zx[(size_t)r * DPROJ + tid]);
  float v1 = y[(size_t)r * DIN + 256 + tid] * siluf(zx[(size_t)r * DPROJ + 256 + tid]);
  float ss = block_sum256(v0 * v0 + v1 * v1, red) * (1.0f / DIN);
  float rs = rsqrtf(ss + EPSV);
  out[(size_t)r * DIN + tid]       = (__bf16)(v0 * rs * rmsw[tid]);
  out[(size_t)r * DIN + 256 + tid] = (__bf16)(v1 * rs * rmsw[256 + tid]);
}

// ---------------- restore + concat + LayerNorm(1536) -> bf16 ----------------
__global__ void __launch_bounds__(256) multi_ln_kernel(
    const float* __restrict__ xs, const int* __restrict__ inv,
    const float* __restrict__ w, const float* __restrict__ b,
    __bf16* __restrict__ out) {
  __shared__ float red[256];
  const int i = blockIdx.x, tid = threadIdx.x;
  float v[NDIR];
#pragma unroll
  for (int d = 0; d < NDIR; d++) {
    int src = inv[(size_t)d * LSEQ + i];
    v[d] = xs[((size_t)d * LSEQ + src) * DMODEL + tid];
  }
  float s = 0.f;
#pragma unroll
  for (int d = 0; d < NDIR; d++) s += v[d];
  float mu = block_sum256(s, red) * (1.0f / (NDIR * DMODEL));
  float q = 0.f;
#pragma unroll
  for (int d = 0; d < NDIR; d++) { float dd = v[d] - mu; q += dd * dd; }
  float var = block_sum256(q, red) * (1.0f / (NDIR * DMODEL));
  float rs = rsqrtf(var + EPSV);
#pragma unroll
  for (int d = 0; d < NDIR; d++)
    out[(size_t)i * (NDIR * DMODEL) + d * DMODEL + tid] =
        (__bf16)((v[d] - mu) * rs * w[d * DMODEL + tid] + b[d * DMODEL + tid]);
}

// ---------------- deterministic mean over L ----------------
__global__ void __launch_bounds__(256) mean_partial_kernel(const float* __restrict__ f,
                                                           float* __restrict__ part) {
  int blk = blockIdx.x, tid = threadIdx.x;
  float s = 0.f;
  for (int r = 0; r < 256; r++) s += f[(size_t)(blk * 256 + r) * DMODEL + tid];
  part[blk * DMODEL + tid] = s;
}
__global__ void __launch_bounds__(256) mean_final_kernel(const float* __restrict__ part,
                                                         float* __restrict__ macc) {
  int tid = threadIdx.x;
  float s = 0.f;
  for (int b2 = 0; b2 < 64; b2++) s += part[b2 * DMODEL + tid];
  macc[tid] = s * (1.0f / LSEQ);
}

// ---------------- classifier head (single block) ----------------
__global__ void __launch_bounds__(256) cls_kernel(
    const float* __restrict__ macc,
    const float* __restrict__ clnw, const float* __restrict__ clnb,
    const float* __restrict__ w1, const float* __restrict__ b1,
    const float* __restrict__ w2, const float* __restrict__ b2,
    const float* __restrict__ w3, const float* __restrict__ b3,
    float* __restrict__ out) {
  __shared__ float red[256];
  __shared__ float hc[768];
  __shared__ float h1[512];
  __shared__ float h2[256];
  int tid = threadIdx.x;
  float mv = macc[tid];
  // pooled = [mean, mean, mean]; LN over 768
  float mu = block_sum256(mv, red) * (1.0f / 256.0f);
  float dv = mv - mu;
  float var = block_sum256(dv * dv, red) * (1.0f / 256.0f);
  float rs = rsqrtf(var + EPSV);
#pragma unroll
  for (int sc = 0; sc < 3; sc++)
    hc[sc * 256 + tid] = dv * rs * clnw[sc * 256 + tid] + clnb[sc * 256 + tid];
  __syncthreads();
  for (int j = tid; j < 512; j += 256) {
    float a = b1[j];
    for (int k = 0; k < 768; k++) a += hc[k] * w1[k * 512 + j];
    h1[j] = geluf(a);
  }
  __syncthreads();
  {
    float a = b2[tid];
    for (int k = 0; k < 512; k++) a += h1[k] * w2[k * 256 + tid];
    h2[tid] = geluf(a);
  }
  __syncthreads();
  if (tid < 11) {
    float a = b3[tid];
    for (int k = 0; k < 256; k++) a += h2[k] * w3[k * 11 + tid];
    out[tid] = a;
  }
}

// ================= host orchestration =================
static inline int ceildiv_i(int a, int b) { return (a + b - 1) / b; }

extern "C" void kernel_launch(void* const* d_in, const int* in_sizes, int n_in,
                              void* d_out, int out_size, void* d_ws, size_t ws_size,
                              hipStream_t stream) {
  (void)in_sizes; (void)n_in; (void)out_size; (void)ws_size;
  const float* vec_real  = (const float*)d_in[0];
  const float* vec_imag  = (const float*)d_in[1];
  const int*   coords    = (const int*)  d_in[2];
  const float* inp_w     = (const float*)d_in[3];
  const float* inp_b     = (const float*)d_in[4];
  const float* inp_ln_w  = (const float*)d_in[5];
  const float* inp_ln_b  = (const float*)d_in[6];
  const float* ln_w      = (const float*)d_in[7];
  const float* ln_b      = (const float*)d_in[8];
  const float* in_proj_w = (const float*)d_in[9];
  const float* conv_w    = (const float*)d_in[10];
  const float* conv_b    = (const float*)d_in[11];
  const float* dt_bias   = (const float*)d_in[12];
  const float* A_log     = (const float*)d_in[13];
  const float* Dskip     = (const float*)d_in[14];
  const float* rms_w     = (const float*)d_in[15];
  const float* out_proj_w= (const float*)d_in[16];
  const float* fus_ln_w  = (const float*)d_in[17];
  const float* fus_ln_b  = (const float*)d_in[18];
  const float* fus_w     = (const float*)d_in[19];
  const float* fus_b     = (const float*)d_in[20];
  const float* cls_ln_w  = (const float*)d_in[21];
  const float* cls_ln_b  = (const float*)d_in[22];
  const float* cls_w1    = (const float*)d_in[23];
  const float* cls_b1    = (const float*)d_in[24];
  const float* cls_w2    = (const float*)d_in[25];
  const float* cls_b2    = (const float*)d_in[26];
  const float* cls_w3    = (const float*)d_in[27];
  const float* cls_b3    = (const float*)d_in[28];

  // -------- workspace arena --------
  char* wsb = (char*)d_ws;
  size_t cur = 0;
  auto alloc = [&](size_t bytes) -> void* {
    void* p = wsb + cur;
    cur += (bytes + 255) & ~(size_t)255;
    return p;
  };
  __bf16*   xcat   = (__bf16*)  alloc((size_t)LSEQ * 128 * 2);
  __bf16*   w_inp  = (__bf16*)  alloc((size_t)128 * DMODEL * 2);
  __bf16*   w_ip   = (__bf16*)  alloc((size_t)NLAYERS * DMODEL * DPROJ * 2);
  __bf16*   w_op   = (__bf16*)  alloc((size_t)NLAYERS * DIN * DMODEL * 2);
  __bf16*   w_fus  = (__bf16*)  alloc((size_t)(NDIR * DMODEL) * DMODEL * 2);
  float*    h      = (float*)   alloc((size_t)LSEQ * DMODEL * 4);
  unsigned* keys   = (unsigned*)alloc((size_t)NDIR * LSEQ * 4);
  int*      perm   = (int*)     alloc((size_t)NDIR * LSEQ * 4);
  int*      inv    = (int*)     alloc((size_t)NDIR * LSEQ * 4);
  float*    xs     = (float*)   alloc((size_t)NDIR * LSEQ * DMODEL * 4);
  __bf16*   xn     = (__bf16*)  alloc((size_t)LSEQ * DMODEL * 2);
  float*    zx     = (float*)   alloc((size_t)LSEQ * DPROJ * 4);   // reused region
  float*    xconv  = (float*)   alloc((size_t)LSEQ * DXBC * 4);
  float*    dtp    = (float*)   alloc((size_t)LSEQ * NHEADS * 4);
  float*    eA     = (float*)   alloc((size_t)LSEQ * NHEADS * 4);
  float*    ylocal = (float*)   alloc((size_t)LSEQ * DIN * 4);
  float*    locals = (float*)   alloc((size_t)NCHUNK * NHEADS * 4096 * 4);
  float*    states = (float*)   alloc((size_t)NCHUNK * NHEADS * 4096 * 4);
  float*    cdecay = (float*)   alloc((size_t)NCHUNK * NHEADS * 4);
  __bf16*   ygate  = (__bf16*)  alloc((size_t)LSEQ * DIN * 2);
  float*    part   = (float*)   alloc((size_t)64 * DMODEL * 4);
  float*    macc   = (float*)   alloc((size_t)DMODEL * 4);
  // aliases into the zx region (used at disjoint phases)
  float*  hpre  = zx;                                              // 16 MB <= 76 MB
  __bf16* multi = (__bf16*)zx;                                     // 50 MB
  float*  fused = (float*)((char*)zx + (size_t)LSEQ * (NDIR*DMODEL) * 2); // +16 MB

  dim3 blk(256);

  // -------- weight / input precision conversion --------
  cvt_bf16_kernel<<<ceildiv_i(128 * DMODEL, 256), blk, 0, stream>>>(inp_w, w_inp, 128 * DMODEL);
  cvt_bf16_kernel<<<ceildiv_i(NLAYERS * DMODEL * DPROJ, 256), blk, 0, stream>>>(in_proj_w, w_ip, NLAYERS * DMODEL * DPROJ);
  cvt_bf16_kernel<<<ceildiv_i(NLAYERS * DIN * DMODEL, 256), blk, 0, stream>>>(out_proj_w, w_op, NLAYERS * DIN * DMODEL);
  cvt_bf16_kernel<<<ceildiv_i(NDIR * DMODEL * DMODEL, 256), blk, 0, stream>>>(fus_w, w_fus, NDIR * DMODEL * DMODEL);
  concat_embed_kernel<<<LSEQ * 128 / 256, blk, 0, stream>>>(vec_real, vec_imag, xcat);

  // -------- embed: GEMM + LN + GELU --------
  gemm_bf16_kernel<<<dim3(LSEQ / 128, ceildiv_i(DMODEL, 64)), blk, 0, stream>>>(
      xcat, w_inp, inp_b, hpre, LSEQ, DMODEL, 128, GF_BIAS);
  ln256_kernel<<<LSEQ, blk, 0, stream>>>(hpre, inp_ln_w, inp_ln_b, h, nullptr, 1);

  // -------- 6 traversal orders (stable lexsort via ranking) --------
  key_kernel<<<NDIR * LSEQ / 256, blk, 0, stream>>>(coords, keys);
  rank_kernel<<<dim3(LSEQ / 256, NDIR), blk, 0, stream>>>(keys, inv, perm);
  gather_kernel<<<(int)((size_t)NDIR * LSEQ * DMODEL / 256), blk, 0, stream>>>(h, perm, xs);

  // -------- layers x directions --------
  for (int l = 0; l < NLAYERS; l++) {
    for (int d = 0; d < NDIR; d++) {
      float* xd = xs + (size_t)d * LSEQ * DMODEL;
      ln256_kernel<<<LSEQ, blk, 0, stream>>>(xd, ln_w + l * DMODEL, ln_b + l * DMODEL,
                                             nullptr, xn, 0);
      gemm_bf16_kernel<<<dim3(LSEQ / 128, ceildiv_i(DPROJ, 64)), blk, 0, stream>>>(
          xn, w_ip + (size_t)l * DMODEL * DPROJ, nullptr, zx, LSEQ, DPROJ, DMODEL, 0);
      dt_kernel<<<LSEQ * NHEADS / 256, blk, 0, stream>>>(zx, dt_bias + l * NHEADS, dtp);
      conv_kernel<<<(int)((size_t)LSEQ * DXBC / 256), blk, 0, stream>>>(
          zx, conv_w + (size_t)l * DXBC * DCONV, conv_b + (size_t)l * DXBC, xconv);
      ssm_intra_kernel<<<dim3(NCHUNK, NHEADS), blk, 0, stream>>>(
          xconv, dtp, A_log + l * NHEADS, Dskip + l * NHEADS, ylocal, locals, cdecay, eA);
      ssm_state_kernel<<<NHEADS, blk, 0, stream>>>(locals, cdecay, states);
      ssm_inter_kernel<<<dim3(NCHUNK, NHEADS), blk, 0, stream>>>(xconv, states, eA, ylocal);
      gate_rms_kernel<<<LSEQ, blk, 0, stream>>>(ylocal, zx, rms_w + (size_t)l * DIN, ygate);
      gemm_bf16_kernel<<<dim3(LSEQ / 128, ceildiv_i(DMODEL, 64)), blk, 0, stream>>>(
          ygate, w_op + (size_t)l * DIN * DMODEL, nullptr, xd, LSEQ, DMODEL, DIN, GF_ACC);
    }
  }

  // -------- fusion: restore + concat + LN -> GEMM + bias + GELU --------
  multi_ln_kernel<<<LSEQ, blk, 0, stream>>>(xs, inv, fus_ln_w, fus_ln_b, multi);
  gemm_bf16_kernel<<<dim3(LSEQ / 128, ceildiv_i(DMODEL, 64)), blk, 0, stream>>>(
      multi, w_fus, fus_b, fused, LSEQ, DMODEL, NDIR * DMODEL, GF_BIAS | GF_GELU);

  // -------- deterministic global mean + classifier --------
  mean_partial_kernel<<<64, blk, 0, stream>>>(fused, part);
  mean_final_kernel<<<1, blk, 0, stream>>>(part, macc);
  cls_kernel<<<1, blk, 0, stream>>>(macc, cls_ln_w, cls_ln_b, cls_w1, cls_b1,
                                    cls_w2, cls_b2, cls_w3, cls_b3, (float*)d_out);
}
